// NERModel_6322191860148
// MI455X (gfx1250) — compile-verified, matrix operand order checked by reading, not run on hardware
//
#include <hip/hip_runtime.h>
#include <hip/hip_bf16.h>
#include <math.h>

typedef __attribute__((ext_vector_type(16))) _Float16 v16h;
typedef __attribute__((ext_vector_type(8)))  _Float16 v8h;
typedef __attribute__((ext_vector_type(8)))  float    v8f;

#define L_SEQ 4096
#define DW 100
#define DI 100
#define KP 224          // DW+DI=200 padded to 7*32
#define NKT 7           // K tiles for input GEMM
#define NG 128          // 4*HD gates
#define HD 32
#define TT 45           // tags
#define NEGC (-100.0f)

// ---------------- embedding concat -> f16 padded X ----------------
__global__ void build_x_kernel(const int* __restrict__ sentence,
                               const int* __restrict__ intent,
                               const float* __restrict__ word_emb,
                               const float* __restrict__ intent_emb,
                               _Float16* __restrict__ Xh) {
    int idx = blockIdx.x * blockDim.x + threadIdx.x;
    if (idx >= L_SEQ * KP) return;
    int t = idx / KP, k = idx % KP;
    float v = 0.f;
    if (k < DW)            v = word_emb[(long)sentence[t] * DW + k];
    else if (k < DW + DI)  v = intent_emb[(long)intent[0] * DI + (k - DW)];
    Xh[idx] = (_Float16)v;
}

// ---------------- weight conversions to f16 (padded) ----------------
__global__ void conv_wih_kernel(const float* __restrict__ wf,
                                const float* __restrict__ wb,
                                _Float16* __restrict__ Wih) {
    int idx = blockIdx.x * blockDim.x + threadIdx.x;       // 2*128*224
    if (idx >= 2 * NG * KP) return;
    int dir = idx / (NG * KP);
    int r   = (idx / KP) % NG;
    int k   = idx % KP;
    const float* w = dir ? wb : wf;
    Wih[idx] = (_Float16)(k < (DW + DI) ? w[r * (DW + DI) + k] : 0.f);
}

__global__ void conv_wout_kernel(const float* __restrict__ W_out,
                                 _Float16* __restrict__ Wout) {
    int idx = blockIdx.x * blockDim.x + threadIdx.x;       // 48*64
    if (idx >= 48 * 64) return;
    int r = idx / 64, k = idx % 64;
    Wout[idx] = (_Float16)(r < TT ? W_out[r * 64 + k] : 0.f);
}

// ---------------- constrained transitions ----------------
__global__ void build_trans_kernel(const float* __restrict__ transitions,
                                   const int* __restrict__ mask,
                                   const int* __restrict__ is_O,
                                   const int* __restrict__ is_I,
                                   float* __restrict__ tc) {
    int idx = blockIdx.x * blockDim.x + threadIdx.x;
    if (idx >= TT * TT) return;
    int i = idx / TT, j = idx % TT;
    bool ai = mask[i] != 0, aj = mask[j] != 0;
    bool pair = ai && aj;
    bool forbid = (!pair) || (pair && (is_O[i] != 0) && (is_I[j] != 0));
    tc[idx] = forbid ? NEGC : transitions[idx];
}

// ---------------- WMMA fragment load (16-bit A/B layout, wave32) ----------------
__device__ __forceinline__ v16h load_frag16(const _Float16* p) {
    // p already offset by (lane>=16 ? 8 : 0) in K; elements 0..7 = K+0..7, 8..15 = K+16..23
    v8h lo = *(const v8h*)(p);
    v8h hi = *(const v8h*)(p + 16);
    v16h r;
#pragma unroll
    for (int i = 0; i < 8; ++i) { r[i] = lo[i]; r[i + 8] = hi[i]; }
    return r;
}

// ---------------- GEMM1: xw[dir][L][128] = Xh[L][224] @ Wih[dir]^T ----------------
__global__ __launch_bounds__(256)
void lstm_ingemm_kernel(const _Float16* __restrict__ Xh,
                        const _Float16* __restrict__ Wih,
                        float* __restrict__ xw) {
    int dir  = blockIdx.y;
    int mt   = blockIdx.x;                 // 0..255  (M tile)
    int wave = threadIdx.x >> 5;           // 0..7    (N tile: 8*16 = 128 gates)
    int lane = threadIdx.x & 31;
    int half = lane >> 4, lm = lane & 15;
    int arow = mt * 16 + lm;               // A row (time step)
    int bcol = wave * 16 + lm;             // B col == w_ih row (gate)
    const _Float16* apB = Xh + (long)arow * KP + half * 8;
    const _Float16* bpB = Wih + ((long)dir * NG + bcol) * KP + half * 8;
    v8f acc = {};
#pragma unroll
    for (int kt = 0; kt < NKT; ++kt) {
        v16h a = load_frag16(apB + kt * 32);
        v16h b = load_frag16(bpB + kt * 32);
        acc = __builtin_amdgcn_wmma_f32_16x16x32_f16(
            false, a, false, b, (short)0, acc, false, false);
    }
    float* outp = xw + ((long)dir * L_SEQ + mt * 16) * NG + wave * 16 + lm;
#pragma unroll
    for (int v = 0; v < 8; ++v) {
        int r = v + 8 * half;              // C/D layout: lanes 0-15 rows 0..7, 16-31 rows 8..15
        outp[(long)r * NG] = acc[v];
    }
}

// ---------------- sequential LSTM recurrence: 1 wave per direction ----------------
__device__ __forceinline__ float sigm(float x) { return 1.f / (1.f + expf(-x)); }

__device__ __forceinline__ float bcast_lane(float v, int lane) {
    return __int_as_float(__builtin_amdgcn_readlane(__float_as_int(v), lane));
}

__global__ __launch_bounds__(32)
void lstm_rec_kernel(const float* __restrict__ xw,
                     const float* __restrict__ w_hh_f, const float* __restrict__ b_f,
                     const float* __restrict__ w_hh_b, const float* __restrict__ b_b,
                     const float* __restrict__ h0, const float* __restrict__ c0,
                     _Float16* __restrict__ hs) {
    int dir = blockIdx.x;
    int j   = threadIdx.x;                 // lane owns gates j, j+32, j+64, j+96
    const float* whh  = dir ? w_hh_b : w_hh_f;
    const float* bias = dir ? b_b : b_f;
    float w0[HD], w1[HD], w2[HD], w3[HD];
#pragma unroll
    for (int i = 0; i < HD; ++i) {
        w0[i] = whh[(j          ) * HD + i];
        w1[i] = whh[(j + HD     ) * HD + i];
        w2[i] = whh[(j + 2 * HD ) * HD + i];
        w3[i] = whh[(j + 3 * HD ) * HD + i];
    }
    float bi = bias[j], bf = bias[j + HD], bg = bias[j + 2 * HD], bo = bias[j + 3 * HD];
    float h = h0[dir * HD + j], c = c0[dir * HD + j];
    const float* xwd = xw + (long)dir * L_SEQ * NG;
    for (int t = 0; t < L_SEQ; ++t) {
        int tt = dir ? (L_SEQ - 1 - t) : t;
        const float* xr = xwd + (long)tt * NG;
        float gi = xr[j] + bi, gf = xr[j + HD] + bf;
        float gg = xr[j + 2 * HD] + bg, go = xr[j + 3 * HD] + bo;
#pragma unroll
        for (int i = 0; i < HD; ++i) {
            // constant lane index -> v_readlane_b32 into SGPR, FMAs use SGPR operand
            float hv = bcast_lane(h, i);
            gi += hv * w0[i]; gf += hv * w1[i];
            gg += hv * w2[i]; go += hv * w3[i];
        }
        c = sigm(gf) * c + sigm(gi) * tanhf(gg);
        h = sigm(go) * tanhf(c);
        hs[(long)tt * 64 + dir * HD + j] = (_Float16)h;
    }
}

// ---------------- emissions GEMM: em[L][45] = hs[L][64] @ Wout^T + b + maskNEG ----------------
__global__ __launch_bounds__(32)
void emis_gemm_kernel(const _Float16* __restrict__ hs,
                      const _Float16* __restrict__ Wout,
                      const float* __restrict__ b_out,
                      const int* __restrict__ mask,
                      float* __restrict__ em) {
    int mt = blockIdx.x;                   // 0..255
    int nt = blockIdx.y;                   // 0..2
    int lane = threadIdx.x & 31;
    int half = lane >> 4, lm = lane & 15;
    int arow = mt * 16 + lm;
    int bcol = nt * 16 + lm;
    const _Float16* apB = hs + (long)arow * 64 + half * 8;
    const _Float16* bpB = Wout + (long)bcol * 64 + half * 8;
    v8f acc = {};
#pragma unroll
    for (int kt = 0; kt < 2; ++kt) {
        v16h a = load_frag16(apB + kt * 32);
        v16h b = load_frag16(bpB + kt * 32);
        acc = __builtin_amdgcn_wmma_f32_16x16x32_f16(
            false, a, false, b, (short)0, acc, false, false);
    }
    int col = nt * 16 + lm;
    if (col < TT) {
        float add = b_out[col] + (mask[col] ? 0.f : NEGC);
#pragma unroll
        for (int v = 0; v < 8; ++v) {
            int r = mt * 16 + v + 8 * half;
            em[(long)r * TT + col] = acc[v] + add;
        }
    }
}

// ---------------- Viterbi scan + backtrace: SINGLE WAVE (barriers -> S_NOP) ----------------
__global__ __launch_bounds__(32)
void viterbi_kernel(const float* __restrict__ em,
                    const float* __restrict__ tc,
                    const float* __restrict__ start_trans,
                    const float* __restrict__ end_trans,
                    int* __restrict__ bps,
                    float* __restrict__ out) {
    __shared__ float s[TT];
    __shared__ float tr[TT * TT];
    int j  = threadIdx.x;                  // col 1: always valid (j < 32 <= 45)
    int j2 = j + 32;                       // col 2: valid when j < 13
    bool has2 = (j2 < TT);
    for (int k = j; k < TT * TT; k += 32) tr[k] = tc[k];
    s[j] = start_trans[j] + em[j];
    if (has2) s[j2] = start_trans[j2] + em[j2];
    __syncthreads();                       // single wave: lowers to waitcnt (no HW barrier)
    for (int t = 1; t < L_SEQ; ++t) {
        float b1 = -3.0e38f, b2 = -3.0e38f;
        int   a1 = 0,        a2 = 0;
        for (int i = 0; i < TT; ++i) {
            float si = s[i];
            float v1 = si + tr[i * TT + j];
            if (v1 > b1) { b1 = v1; a1 = i; }
            if (has2) {
                float v2 = si + tr[i * TT + j2];
                if (v2 > b2) { b2 = v2; a2 = i; }
            }
        }
        bps[(long)(t - 1) * TT + j] = a1;
        if (has2) bps[(long)(t - 1) * TT + j2] = a2;
        __syncthreads();
        s[j] = b1 + em[(long)t * TT + j];
        if (has2) s[j2] = b2 + em[(long)t * TT + j2];
        __syncthreads();
    }
    __threadfence();
    __syncthreads();
    if (j == 0) {
        float bs = -3.0e38f; int last = 0;
        for (int i = 0; i < TT; ++i) {
            float v = s[i] + end_trans[i];
            if (v > bs) { bs = v; last = i; }
        }
        out[L_SEQ] = bs;
        out[L_SEQ - 1] = (float)last;
        int cur = last;
        for (int t = L_SEQ - 1; t >= 1; --t) {
            cur = bps[(long)(t - 1) * TT + cur];
            out[t - 1] = (float)cur;
        }
    }
}

// ---------------- host launcher ----------------
extern "C" void kernel_launch(void* const* d_in, const int* in_sizes, int n_in,
                              void* d_out, int out_size, void* d_ws, size_t ws_size,
                              hipStream_t stream) {
    (void)in_sizes; (void)n_in; (void)out_size; (void)ws_size;
    const int*   sentence   = (const int*)  d_in[0];
    const int*   intent     = (const int*)  d_in[1];
    const int*   mask       = (const int*)  d_in[2];
    const float* word_emb   = (const float*)d_in[3];
    const float* intent_emb = (const float*)d_in[4];
    const float* w_ih_f     = (const float*)d_in[5];
    const float* w_hh_f     = (const float*)d_in[6];
    const float* b_f        = (const float*)d_in[7];
    const float* w_ih_b     = (const float*)d_in[8];
    const float* w_hh_b     = (const float*)d_in[9];
    const float* b_b        = (const float*)d_in[10];
    const float* h0         = (const float*)d_in[11];
    const float* c0         = (const float*)d_in[12];
    const float* W_out      = (const float*)d_in[13];
    const float* b_out      = (const float*)d_in[14];
    const float* transitions= (const float*)d_in[15];
    const float* start_tr   = (const float*)d_in[16];
    const float* end_tr     = (const float*)d_in[17];
    const int*   is_O       = (const int*)  d_in[18];
    const int*   is_I       = (const int*)  d_in[19];
    float* out = (float*)d_out;

    // workspace carve-up (256B aligned)
    char* ws = (char*)d_ws;
    size_t off = 0;
    auto carve = [&](size_t bytes) { char* p = ws + off; off = (off + bytes + 255) & ~(size_t)255; return p; };
    _Float16* Xh   = (_Float16*)carve((size_t)L_SEQ * KP * 2);       // 1.75 MB
    _Float16* Wih  = (_Float16*)carve((size_t)2 * NG * KP * 2);      // 112 KB
    _Float16* Wout = (_Float16*)carve((size_t)48 * 64 * 2);          // 6 KB
    float*    xw   = (float*)   carve((size_t)2 * L_SEQ * NG * 4);   // 4 MB
    _Float16* hs   = (_Float16*)carve((size_t)L_SEQ * 64 * 2);       // 512 KB
    float*    em   = (float*)   carve((size_t)L_SEQ * TT * 4);       // 720 KB
    float*    tc   = (float*)   carve((size_t)TT * TT * 4);          // 8 KB
    int*      bps  = (int*)     carve((size_t)(L_SEQ - 1) * TT * 4); // 720 KB

    // 1. inputs -> f16 padded X
    build_x_kernel<<<(L_SEQ * KP + 255) / 256, 256, 0, stream>>>(
        sentence, intent, word_emb, intent_emb, Xh);
    // 2-4. weight conversions + constrained transitions
    conv_wih_kernel<<<(2 * NG * KP + 255) / 256, 256, 0, stream>>>(w_ih_f, w_ih_b, Wih);
    conv_wout_kernel<<<(48 * 64 + 255) / 256, 256, 0, stream>>>(W_out, Wout);
    build_trans_kernel<<<(TT * TT + 255) / 256, 256, 0, stream>>>(
        transitions, mask, is_O, is_I, tc);
    // 5. big WMMA GEMM: input projections for both directions
    lstm_ingemm_kernel<<<dim3(L_SEQ / 16, 2), 256, 0, stream>>>(Xh, Wih, xw);
    // 6. sequential recurrence, one wave per direction
    lstm_rec_kernel<<<2, 32, 0, stream>>>(xw, w_hh_f, b_f, w_hh_b, b_b, h0, c0, hs);
    // 7. WMMA emissions GEMM
    emis_gemm_kernel<<<dim3(L_SEQ / 16, 3), 32, 0, stream>>>(hs, Wout, b_out, mask, em);
    // 8. Viterbi scan + backtrace (single wave, no HW barriers)
    viterbi_kernel<<<1, 32, 0, stream>>>(em, tc, start_tr, end_tr, bps, out);
}
